// ReluLSTM_49263274885462
// MI455X (gfx1250) — matrix operand order repaired and checked by reading, not compiled, not run on hardware
//
#include <hip/hip_runtime.h>
#include <hip/hip_bf16.h>

// ---------------------------------------------------------------------------
// LSTM on MI455X (gfx1250): bf16 WMMA recurrence with async global->LDS
// double buffering.
//   gates[64,4096] = [h;x][64,2048] @ W[2048,4096] + b   (per step, 512 steps)
// One wave owns one 16x16 hidden tile and accumulates f,i,g,o gate tiles
// (4 WMMA accumulators) -> in-register cell update.
// W pre-packed once into wave32 B-fragment layout; resident in 192MB L2.
// Hot loop: k-block kb+2 streams L2->LDS via GLOBAL_LOAD_ASYNC_TO_LDS_B128
// (ASYNCcnt) while kb's fragments are pulled LDS->VGPR via ds_load_b128
// (DScnt) and fed to 4 WMMAs. The L2 round-trip is throttled explicitly with
// s_wait_asynccnt; compiler-inserted waits in the loop are cheap dscnt waits.
// ---------------------------------------------------------------------------

typedef __attribute__((ext_vector_type(16))) __bf16   v16bf;
typedef __attribute__((ext_vector_type(8)))  float    v8f;
typedef __attribute__((ext_vector_type(4)))  unsigned u32x4;
typedef __attribute__((ext_vector_type(4)))  int      v4i;

union B16Frag { v16bf v; u32x4 q[2]; };

#define HSZ   1024
#define ISZ   1024
#define BATCH 64
#define SEQ   512
#define GDIM  4096          // 4*HSZ
#define KTOT  2048          // HSZ + ISZ
#define NB    256           // GDIM / 16

// LDS staging: per wave per stage: A (32 lanes * 32B = 1KB) + 4 gate B frags
// (4 * 1KB). Elements (uint16): 512 (A) + 2048 (B) = 2560. Two stages.
#define STAGE_ELEMS 2560
#define WAVE_STAGE_OFF(s, wid) ((s) * 8 * STAGE_ELEMS + (wid) * STAGE_ELEMS)
#define LDS_BYTES (2 * 8 * STAGE_ELEMS * 2)   // 81920

typedef __attribute__((address_space(1))) v4i            glb_v4i;   // global
typedef __attribute__((address_space(3))) v4i            lds_v4i;   // LDS
typedef __attribute__((address_space(3))) unsigned short lds_u16;
typedef __attribute__((address_space(3))) const u32x4    lds_cu32x4;

static __device__ __forceinline__ unsigned short f32_to_bf16(float f) {
    unsigned u = __float_as_uint(f);
    u += 0x7FFFu + ((u >> 16) & 1u);      // round-to-nearest-even
    return (unsigned short)(u >> 16);
}

// Pack W[2048,4096] fp32 row-major -> bf16 wave32 B-fragment layout:
// packed[((kb*NB + nb)*512) + lane*16 + j]
//   lane<16 : element j holds W[kb*32 + j      , nb*16 + lane   ]
//   lane>=16: element j holds W[kb*32 + 16 + j , nb*16 + lane-16]
__global__ void lstm_pack_w(const float* __restrict__ W,
                            unsigned short* __restrict__ packed) {
    int idx = blockIdx.x * blockDim.x + threadIdx.x;
    if (idx >= KTOT * GDIM) return;
    int kb   = idx >> 17;          // / (NB*512)
    int rem  = idx & 131071;
    int nb   = rem >> 9;
    int r2   = rem & 511;
    int lane = r2 >> 4;
    int j    = r2 & 15;
    int k = (kb << 5) + ((lane >> 4) << 4) + j;
    int n = (nb << 4) + (lane & 15);
    packed[idx] = f32_to_bf16(W[(size_t)k * GDIM + n]);
}

__global__ void lstm_cvt_x(const float* __restrict__ X,
                           unsigned short* __restrict__ Xb, int n) {
    int idx = blockIdx.x * blockDim.x + threadIdx.x;
    if (idx < n) Xb[idx] = f32_to_bf16(X[idx]);
}

__global__ void lstm_init_hc(const float* __restrict__ h0,
                             const float* __restrict__ c0,
                             unsigned short* __restrict__ h,
                             float* __restrict__ c) {
    int idx = blockIdx.x * blockDim.x + threadIdx.x;   // 64*1024
    if (idx >= BATCH * HSZ) return;
    int j = idx & (HSZ - 1);
    h[idx] = f32_to_bf16(h0[j]);
    c[idx] = c0[j];
}

// 16-byte async copy: global -> LDS, tracked by ASYNCcnt.
static __device__ __forceinline__ void
async_cp16(const unsigned short* gsrc, unsigned short* ldst) {
    __builtin_amdgcn_global_load_async_to_lds_b128(
        (glb_v4i*)gsrc, (lds_v4i*)ldst, 0, 0);
}

struct Frags { B16Frag a, bF, bI, bG, bO; };

static __device__ __forceinline__ u32x4 lds_read16(const unsigned short* p) {
    return *(lds_cu32x4*)((lds_u16*)p);
}

// Issue the 10 async b128 copies that stage k-block kb for this wave.
static __device__ __forceinline__ void
prefetch_block(int kb, unsigned short* wbase, int lane,
               const unsigned short* __restrict__ hrow,
               const unsigned short* __restrict__ xrow,
               const unsigned short* __restrict__ bb0) {
    const unsigned short* asrc =
        (kb < 32) ? (hrow + kb * 32) : (xrow + (kb - 32) * 32);
    unsigned short* aDst = wbase + lane * 16;
    async_cp16(asrc,     aDst);
    async_cp16(asrc + 8, aDst + 8);

    const unsigned short* bsrc = bb0 + (size_t)kb * 131072;
    unsigned short* bDst = wbase + 512 + lane * 16;
    #pragma unroll
    for (int g = 0; g < 4; ++g) {
        async_cp16(bsrc + g * 32768,     bDst + g * 512);
        async_cp16(bsrc + g * 32768 + 8, bDst + g * 512 + 8);
    }
}

static __device__ __forceinline__ Frags
load_frags_lds(const unsigned short* wbase, int lane) {
    Frags f;
    const unsigned short* ap = wbase + lane * 16;
    f.a.q[0]  = lds_read16(ap);
    f.a.q[1]  = lds_read16(ap + 8);
    const unsigned short* bp = wbase + 512 + lane * 16;
    f.bF.q[0] = lds_read16(bp);        f.bF.q[1] = lds_read16(bp + 8);
    f.bI.q[0] = lds_read16(bp + 512);  f.bI.q[1] = lds_read16(bp + 520);
    f.bG.q[0] = lds_read16(bp + 1024); f.bG.q[1] = lds_read16(bp + 1032);
    f.bO.q[0] = lds_read16(bp + 1536); f.bO.q[1] = lds_read16(bp + 1544);
    return f;
}

static __device__ __forceinline__ void
do_wmma(const Frags& f, v8f& accF, v8f& accI, v8f& accG, v8f& accO) {
    accF = __builtin_amdgcn_wmma_f32_16x16x32_bf16(false, f.a.v, false, f.bF.v,
                                                   (short)0, accF, false, false);
    accI = __builtin_amdgcn_wmma_f32_16x16x32_bf16(false, f.a.v, false, f.bI.v,
                                                   (short)0, accI, false, false);
    accG = __builtin_amdgcn_wmma_f32_16x16x32_bf16(false, f.a.v, false, f.bG.v,
                                                   (short)0, accG, false, false);
    accO = __builtin_amdgcn_wmma_f32_16x16x32_bf16(false, f.a.v, false, f.bO.v,
                                                   (short)0, accO, false, false);
}

// One recurrence step. Grid: 32 blocks x 256 threads = 256 waves.
// wave -> (mt 0..3, nt 0..63): rows [mt*16,mt*16+16), hidden cols [nt*16,nt*16+16)
__global__ void __launch_bounds__(256, 1)
lstm_step(const unsigned short* __restrict__ hprev,
          const float*          __restrict__ cprev,
          unsigned short*       __restrict__ hnext,
          float*                __restrict__ cnext,
          const unsigned short* __restrict__ xt,       // [64,1024] bf16
          const unsigned short* __restrict__ packedW,  // B-fragment layout
          const float*          __restrict__ bias,     // [4096]
          float* __restrict__ outH, float* __restrict__ outC) {
    extern __shared__ unsigned short lds[];

    const int tid  = threadIdx.x;
    const int wid  = tid >> 5;
    const int wave = blockIdx.x * (blockDim.x >> 5) + wid;
    const int lane = tid & 31;
    const int mt   = wave >> 6;            // 0..3
    const int nt   = wave & 63;            // 0..63
    const int m0   = mt << 4;
    const int col  = lane & 15;
    const int hi   = lane >> 4;            // 0 or 1

    v8f accF = {0.f,0.f,0.f,0.f,0.f,0.f,0.f,0.f};
    v8f accI = accF, accG = accF, accO = accF;

    // A layout: lanes 0-15 and 16-31 both hold M=0..15; hi lanes are offset
    // +16 in K within each 32-wide K block.
    const int arow = m0 + col;
    const unsigned short* hrow = hprev + (size_t)arow * HSZ + hi * 16;
    const unsigned short* xrow = xt    + (size_t)arow * ISZ + hi * 16;

    // B base: packed + kb*131072 + nb*512 + lane*16 ; gates step nb by 64.
    const unsigned short* bb0 = packedW + (size_t)nt * 512 + (size_t)lane * 16;

    unsigned short* stage0 = lds + WAVE_STAGE_OFF(0, wid);
    unsigned short* stage1 = lds + WAVE_STAGE_OFF(1, wid);

    prefetch_block(0, stage0, lane, hrow, xrow, bb0);
    prefetch_block(1, stage1, lane, hrow, xrow, bb0);

    #pragma unroll
    for (int kb = 0; kb < 64; ++kb) {
        unsigned short* st = (kb & 1) ? stage1 : stage0;
        // Let the newest 10-instruction prefetch stay in flight; drain on the
        // final iterations where nothing further is prefetched.
        if (kb < 62) asm volatile("s_wait_asynccnt 0xa" ::: "memory");
        else         asm volatile("s_wait_asynccnt 0x0" ::: "memory");
        Frags f = load_frags_lds(st, lane);
        do_wmma(f, accF, accI, accG, accO);
        if (kb + 2 < 64)
            prefetch_block(kb + 2, st, lane, hrow, xrow, bb0);
    }

    // Bias: every element of a lane's C fragment shares one column.
    const int ncol = (nt << 4) + col;          // hidden index 0..1023
    const float bF = bias[ncol];
    const float bI = bias[ncol + 1024];
    const float bG = bias[ncol + 2048];
    const float bO = bias[ncol + 3072];

    // C layout: element r -> row m0 + r + 8*hi, column ncol.
    #pragma unroll
    for (int r = 0; r < 8; ++r) {
        const int row = m0 + r + (hi << 3);
        const int idx = row * HSZ + ncol;
        const float fv = 1.f / (1.f + __expf(-(accF[r] + bF)));
        const float iv = 1.f / (1.f + __expf(-(accI[r] + bI)));
        const float gv = tanhf(accG[r] + bG);
        const float ov = 1.f / (1.f + __expf(-(accO[r] + bO)));
        const float cn = fv * cprev[idx] + iv * gv;
        const float hn = ov * tanhf(cn);
        cnext[idx] = cn;
        hnext[idx] = f32_to_bf16(hn);
        if (outH) { outH[idx] = hn; outC[idx] = cn; }
    }
}

extern "C" void kernel_launch(void* const* d_in, const int* in_sizes, int n_in,
                              void* d_out, int out_size, void* d_ws, size_t ws_size,
                              hipStream_t stream) {
    (void)in_sizes; (void)n_in; (void)out_size; (void)ws_size;
    const float* X  = (const float*)d_in[0];   // [512,64,1024]
    const float* W  = (const float*)d_in[1];   // [2048,4096]
    const float* b  = (const float*)d_in[2];   // [4096]
    const float* h0 = (const float*)d_in[3];   // [1024]
    const float* c0 = (const float*)d_in[4];   // [1024]
    float* out = (float*)d_out;                // [h(65536) | c(65536)]

    unsigned char* ws = (unsigned char*)d_ws;
    unsigned short* packedW = (unsigned short*)ws;                        // 16 MB
    unsigned short* Xb      = (unsigned short*)(ws + (size_t)(16 << 20)); // 64 MB
    size_t off = (size_t)(16 << 20) + (size_t)(64 << 20);
    unsigned short* hbuf[2] = { (unsigned short*)(ws + off),
                                (unsigned short*)(ws + off + 131072) };
    off += 262144;
    float* cbuf[2] = { (float*)(ws + off), (float*)(ws + off + 262144) };

    lstm_pack_w<<<(KTOT * GDIM) / 256, 256, 0, stream>>>(W, packedW);
    const int nx = SEQ * BATCH * ISZ;
    lstm_cvt_x<<<nx / 256, 256, 0, stream>>>(X, Xb, nx);
    lstm_init_hc<<<(BATCH * HSZ) / 256, 256, 0, stream>>>(h0, c0, hbuf[0], cbuf[0]);

    for (int t = 0; t < SEQ; ++t) {
        const int cur = t & 1, nxt = cur ^ 1;
        const bool last = (t == SEQ - 1);
        lstm_step<<<32, 256, LDS_BYTES, stream>>>(
            hbuf[cur], cbuf[cur], hbuf[nxt], cbuf[nxt],
            Xb + (size_t)t * BATCH * ISZ, packedW, b,
            last ? out : nullptr, last ? (out + BATCH * HSZ) : nullptr);
    }
}
